// LSTMDecoder_86981677679219
// MI455X (gfx1250) — matrix operand order repaired and checked by reading, not compiled, not run on hardware
//
#include <hip/hip_runtime.h>

#define Bdim 64
#define Sdim 512
#define Ddim 512
#define Hdim 512
#define GH   2048  // 4*H

typedef __bf16 v16bf __attribute__((ext_vector_type(16)));
typedef __bf16 v8bf  __attribute__((ext_vector_type(8)));
typedef float  v8f   __attribute__((ext_vector_type(8)));

__device__ __forceinline__ unsigned short f2b(float f) {
  unsigned u = __float_as_uint(f);
  u += 0x7FFFu + ((u >> 16) & 1u);  // round-to-nearest-even
  return (unsigned short)(u >> 16);
}

__device__ __forceinline__ v8bf ld8(const unsigned short* p) {
  v8bf v;
  __builtin_memcpy(&v, p, 16);
  return v;
}

__device__ __forceinline__ v16bf combine16(v8bf lo, v8bf hi) {
  v16bf r;
#pragma unroll
  for (int i = 0; i < 8; ++i) { r[i] = lo[i]; r[i + 8] = hi[i]; }
  return r;
}

// A fragment (16x32 bf16, M x K), source row-major.
// lane<16: row=lane,   K = {k0..k0+7, k0+16..k0+23}
// lane>=16: row=lane-16, K = {k0+8..k0+15, k0+24..k0+31}
__device__ __forceinline__ v16bf load_a_frag(const unsigned short* rowPtr, int k0, int hi) {
  const unsigned short* p = rowPtr + k0 + hi * 8;
  return combine16(ld8(p), ld8(p + 16));
}

// B fragment (32x16 bf16, K x N) with B[k][n] = W[n][k] (W row-major).
// lane<16: col=lane, K = {k0..k0+7, k0+16..k0+23}; lane>=16: col=lane-16, K = {k0+8..k0+15, k0+24..k0+31}
__device__ __forceinline__ v16bf load_b_frag(const unsigned short* wRowPtr, int k0, int hi) {
  const unsigned short* p = wRowPtr + k0 + hi * 8;
  return combine16(ld8(p), ld8(p + 16));
}

__device__ __forceinline__ float sigmoidf_(float x) { return 1.0f / (1.0f + __expf(-x)); }

// ---------------- prep kernels ----------------

__global__ void k_prep_weights(const float* __restrict__ wih, const float* __restrict__ whh,
                               const float* __restrict__ bih, const float* __restrict__ bhh,
                               unsigned short* __restrict__ wihb, unsigned short* __restrict__ whhb,
                               float* __restrict__ bias) {
  int i = blockIdx.x * blockDim.x + threadIdx.x;
  if (i < GH * Ddim) { wihb[i] = f2b(wih[i]); whhb[i] = f2b(whh[i]); }
  if (i < GH) bias[i] = bih[i] + bhh[i];
}

__global__ void k_prep_state(const float* __restrict__ hn, const float* __restrict__ cn,
                             unsigned short* __restrict__ hb0, float* __restrict__ cs,
                             unsigned int* __restrict__ bar) {
  int i = blockIdx.x * blockDim.x + threadIdx.x;
  if (i < Bdim * Hdim) { hb0[i] = f2b(hn[i]); cs[i] = cn[i]; }
  if (i == 0) *bar = 0u;
}

__global__ void k_prep_inputs(const float* __restrict__ x, const float* __restrict__ emb,
                              unsigned short* __restrict__ inb) {
  int i = blockIdx.x * blockDim.x + threadIdx.x;
  if (i >= Bdim * Sdim * Ddim) return;
  int d = i % Ddim;
  int s = (i / Ddim) % Sdim;
  int b = i / (Ddim * Sdim);
  float v = (s == 0) ? emb[b * Ddim + d] : x[((size_t)b * Sdim + (s - 1)) * Ddim + d];
  inb[i] = f2b(v);
}

// ---------------- big GEMM: xg[BS, 4H] = inputs @ W_ih^T + bias ----------------
// one wave per (16-row m-tile) x (64-col n-quad); loads for all 4 WMMAs issued
// before the WMMA group so the scheduler can emit partial loadcnt waits.

__global__ void __launch_bounds__(128) k_gemm_xg(const unsigned short* __restrict__ inb,
                                                 const unsigned short* __restrict__ wih,
                                                 const float* __restrict__ bias,
                                                 float* __restrict__ xg) {
  const int lane = threadIdx.x & 31;
  const int wave = threadIdx.x >> 5;
  const int waveId = blockIdx.x * 4 + wave;
  const int mt = waveId >> 5;  // 0..2047
  const int nq = waveId & 31;  // 0..31 (64 columns each)
  const int hi = lane >> 4;
  const int n15 = lane & 15;

  const unsigned short* arow = inb + (size_t)(mt * 16 + n15) * Ddim;
  const unsigned short* wrow0 = wih + (size_t)(nq * 64 + 0 * 16 + n15) * Ddim;
  const unsigned short* wrow1 = wih + (size_t)(nq * 64 + 1 * 16 + n15) * Ddim;
  const unsigned short* wrow2 = wih + (size_t)(nq * 64 + 2 * 16 + n15) * Ddim;
  const unsigned short* wrow3 = wih + (size_t)(nq * 64 + 3 * 16 + n15) * Ddim;

  v8f acc[4];
#pragma unroll
  for (int q = 0; q < 4; ++q) {
    const float bv = bias[nq * 64 + q * 16 + n15];
#pragma unroll
    for (int r = 0; r < 8; ++r) acc[q][r] = bv;
  }

#pragma unroll 4
  for (int k0 = 0; k0 < Ddim; k0 += 32) {
    v16bf a  = load_a_frag(arow, k0, hi);
    v16bf b0 = load_b_frag(wrow0, k0, hi);
    v16bf b1 = load_b_frag(wrow1, k0, hi);
    v16bf b2 = load_b_frag(wrow2, k0, hi);
    v16bf b3 = load_b_frag(wrow3, k0, hi);
    acc[0] = __builtin_amdgcn_wmma_f32_16x16x32_bf16(false, a, false, b0, (short)0, acc[0], false, false);
    acc[1] = __builtin_amdgcn_wmma_f32_16x16x32_bf16(false, a, false, b1, (short)0, acc[1], false, false);
    acc[2] = __builtin_amdgcn_wmma_f32_16x16x32_bf16(false, a, false, b2, (short)0, acc[2], false, false);
    acc[3] = __builtin_amdgcn_wmma_f32_16x16x32_bf16(false, a, false, b3, (short)0, acc[3], false, false);
  }

  const int rbase = mt * 16 + hi * 8;
#pragma unroll
  for (int q = 0; q < 4; ++q) {
    const int col = nq * 64 + q * 16 + n15;
#pragma unroll
    for (int r = 0; r < 8; ++r) xg[(size_t)(rbase + r) * GH + col] = acc[q][r];
  }
}

// ---------------- persistent recurrent kernel ----------------
// 128 blocks x 128 threads. block = (mt batch-tile) x (jt h-col-tile).
// W_hh fragments for each wave's gate tile live in 128 VGPRs for the whole
// kernel (zero B traffic in the hot loop). The shared 16-row h stripe is
// staged into LDS once per step (one clause of 8 b128 loads per thread),
// then the fully unrolled 16-WMMA chain feeds from ds_load_b128.

__global__ void __launch_bounds__(128) k_lstm(const float* __restrict__ xg,
                                              const unsigned short* __restrict__ whh,
                                              unsigned short* hb0, unsigned short* hb1,
                                              float* __restrict__ cs, float* __restrict__ out,
                                              unsigned int* bar) {
  __shared__ unsigned short hlds[16 * Hdim];  // 16 KB: this block's batch-row stripe of h
  __shared__ float gt[4][16][16];             // 4 KB: gate tiles

  const int lane = threadIdx.x & 31;
  const int gate = threadIdx.x >> 5;
  const int mt = blockIdx.x & 3;
  const int jt = blockIdx.x >> 2;
  const int hi = lane >> 4;
  const int n15 = lane & 15;
  const int col0 = gate * Hdim + jt * 16;  // column base in 4H (torch order i,f,g,o)
  const int rbase = mt * 16 + hi * 8;

  // Preload all 16 B fragments (W_hh rows col0+n15, k=0..511) into registers.
  const unsigned short* wrow = whh + (size_t)(col0 + n15) * Hdim;
  v16bf bfrag[16];
#pragma unroll
  for (int kk = 0; kk < 16; ++kk) bfrag[kk] = load_b_frag(wrow, kk * 32, hi);

  const int srow = threadIdx.x >> 3;        // 0..15: stripe row this thread stages
  const int scol = (threadIdx.x & 7) * 64;  // 64-element chunk within the row

  unsigned int target = 0;
  for (int t = 0; t < Sdim; ++t) {
    const unsigned short* hread = (t & 1) ? hb1 : hb0;
    unsigned short* hwrite = (t & 1) ? hb0 : hb1;

    // Stage h stripe -> LDS (issue global clause first).
    {
      const unsigned short* src = hread + (size_t)(mt * 16 + srow) * Hdim + scol;
      unsigned short* dst = &hlds[srow * Hdim + scol];
#pragma unroll
      for (int u = 0; u < 8; ++u) {
        v8bf v = ld8(src + u * 8);
        __builtin_memcpy(dst + u * 8, &v, 16);
      }
    }

    // Seed accumulator with xg[:, t, col] (includes both biases).
    v8f acc;
#pragma unroll
    for (int r = 0; r < 8; ++r)
      acc[r] = xg[((size_t)(rbase + r) * Sdim + t) * GH + col0 + n15];

    __syncthreads();

#pragma unroll
    for (int kk = 0; kk < 16; ++kk) {
      const unsigned short* p = &hlds[n15 * Hdim + kk * 32 + hi * 8];
      v16bf a = combine16(ld8(p), ld8(p + 16));
      acc = __builtin_amdgcn_wmma_f32_16x16x32_bf16(false, a, false, bfrag[kk], (short)0, acc,
                                                    false, false);
    }

#pragma unroll
    for (int r = 0; r < 8; ++r) gt[gate][hi * 8 + r][n15] = acc[r];

    __syncthreads();

    // Block-local cell update: 256 elements, 2 per thread.
    for (int e = threadIdx.x; e < 256; e += 128) {
      const int row = e >> 4, col = e & 15;
      const int b = mt * 16 + row, j = jt * 16 + col;
      const float iv = sigmoidf_(gt[0][row][col]);
      const float fv = sigmoidf_(gt[1][row][col]);
      const float gv = tanhf(gt[2][row][col]);
      const float ov = sigmoidf_(gt[3][row][col]);
      float c = fv * cs[b * Hdim + j] + iv * gv;
      cs[b * Hdim + j] = c;
      float h = ov * tanhf(c);
      out[((size_t)b * Sdim + t) * Hdim + j] = h;
      hwrite[b * Hdim + j] = f2b(h);
    }

    // Grid-wide barrier (monotonic counter); its syncthreads also fence LDS reuse.
    target += gridDim.x;
    __threadfence();
    __syncthreads();
    if (threadIdx.x == 0) {
      __hip_atomic_fetch_add(bar, 1u, __ATOMIC_RELEASE, __HIP_MEMORY_SCOPE_AGENT);
      while (__hip_atomic_load(bar, __ATOMIC_ACQUIRE, __HIP_MEMORY_SCOPE_AGENT) < target)
        __builtin_amdgcn_s_sleep(1);
    }
    __syncthreads();
    __threadfence();
  }
}

// ---------------- launch ----------------

extern "C" void kernel_launch(void* const* d_in, const int* in_sizes, int n_in,
                              void* d_out, int out_size, void* d_ws, size_t ws_size,
                              hipStream_t stream) {
  const float* x   = (const float*)d_in[0];
  const float* emb = (const float*)d_in[1];
  const float* hn  = (const float*)d_in[2];
  const float* cn  = (const float*)d_in[3];
  const float* Wih = (const float*)d_in[4];
  const float* Whh = (const float*)d_in[5];
  const float* bih = (const float*)d_in[6];
  const float* bhh = (const float*)d_in[7];
  float* out = (float*)d_out;

  char* ws = (char*)d_ws;
  size_t off = 0;
  float* xg = (float*)(ws + off);                      off += (size_t)Bdim * Sdim * GH * 4;
  unsigned short* inb  = (unsigned short*)(ws + off);  off += (size_t)Bdim * Sdim * Ddim * 2;
  unsigned short* wihb = (unsigned short*)(ws + off);  off += (size_t)GH * Ddim * 2;
  unsigned short* whhb = (unsigned short*)(ws + off);  off += (size_t)GH * Hdim * 2;
  float* bias = (float*)(ws + off);                    off += (size_t)GH * 4;
  unsigned short* hb0 = (unsigned short*)(ws + off);   off += (size_t)Bdim * Hdim * 2;
  unsigned short* hb1 = (unsigned short*)(ws + off);   off += (size_t)Bdim * Hdim * 2;
  float* cs = (float*)(ws + off);                      off += (size_t)Bdim * Hdim * 4;
  unsigned int* bar = (unsigned int*)(ws + off);       off += 256;

  hipLaunchKernelGGL(k_prep_weights, dim3((GH * Ddim + 255) / 256), dim3(256), 0, stream,
                     Wih, Whh, bih, bhh, wihb, whhb, bias);
  hipLaunchKernelGGL(k_prep_state, dim3((Bdim * Hdim + 255) / 256), dim3(256), 0, stream,
                     hn, cn, hb0, cs, bar);
  hipLaunchKernelGGL(k_prep_inputs, dim3((Bdim * Sdim * Ddim + 255) / 256), dim3(256), 0, stream,
                     x, emb, inb);
  hipLaunchKernelGGL(k_gemm_xg, dim3(16384), dim3(128), 0, stream, inb, wihb, bias, xg);
  hipLaunchKernelGGL(k_lstm, dim3(128), dim3(128), 0, stream, xg, whhb, hb0, hb1, cs, out, bar);
}
